// MultiHeadAttention_63307817943774
// MI455X (gfx1250) — matrix operand order repaired and checked by reading, compiled, not run on hardware
//
#include <hip/hip_runtime.h>
#include <hip/hip_bf16.h>

// ---------------------------------------------------------------------------
// MI455X (gfx1250) fused multi-head attention, wave32 + WMMA f16->f32.
// B=8, S=1024, D=1024, H=16, DK=DV=64.
// ---------------------------------------------------------------------------

typedef __attribute__((ext_vector_type(16))) _Float16 v16h;
typedef __attribute__((ext_vector_type(8)))  float    v8f;
typedef __attribute__((ext_vector_type(4)))  unsigned int uvec4;
typedef __attribute__((ext_vector_type(2)))  unsigned int uvec2;

union Frag16 {
    v16h  h;
    uvec4 u[2];
};
union H2 { _Float16 h[2]; unsigned u; };
union H4 { _Float16 h[4]; uvec2   u; };

__device__ inline v8f zero_v8f() {
    v8f z;
#pragma unroll
    for (int i = 0; i < 8; ++i) z[i] = 0.0f;
    return z;
}

__device__ inline v8f wmma_f16(const Frag16& a, const Frag16& b, v8f c) {
    return __builtin_amdgcn_wmma_f32_16x16x32_f16(
        /*neg_a=*/false, a.h, /*neg_b=*/false, b.h,
        /*c_mod=*/(short)0, c, /*reuse_a=*/false, /*reuse_b=*/false);
}

// ---------------------------------------------------------------------------
// Generic GEMM:  C[M,N] = A[M,K] @ W[K,N] + bias[N]
//   A_HALF:   A is _Float16 (row-major) instead of float.
//   HEAD_OUT: write C as f16 head-major [B=8,H=16,S=1024,64]
//             (row = b*1024+s, col = h*64+d); else f32 row-major.
// Block: 256 threads = 8 waves; tile BM=128 x BN=128, BK=32.
// Wave w computes rows [w*16, w*16+16) x all 128 cols -> 8 WMMA accumulators.
// ---------------------------------------------------------------------------
template <bool A_HALF, bool HEAD_OUT>
__global__ __launch_bounds__(256) void gemm_wmma_kernel(
    const void* __restrict__ Aptr, const float* __restrict__ W,
    const float* __restrict__ bias, void* __restrict__ Cptr,
    int M, int N, int K)
{
    constexpr int BM = 128, BN = 128, BK = 32;
    constexpr int LDA = 40;   // halves; 80B row stride -> 16B-aligned frags
    constexpr int LDB = 40;

    __shared__ _Float16 As[BM * LDA];     // A tile, row-major [row][k]
    __shared__ _Float16 Bs[BN * LDB];     // W tile, col-major [col][k]

    const int tid  = threadIdx.x;
    const int wave = tid >> 5;
    const int lane = tid & 31;
    const int m0   = blockIdx.y * BM;
    const int n0   = blockIdx.x * BN;

    v8f acc[8];
#pragma unroll
    for (int j = 0; j < 8; ++j) acc[j] = zero_v8f();

    for (int k0 = 0; k0 < K; k0 += BK) {
        __syncthreads();   // protect LDS reuse across iterations

        // ---- stage A tile (convert to f16 if needed) ----
        if (A_HALF) {
            const _Float16* A = (const _Float16*)Aptr;
#pragma unroll
            for (int i = 0; i < 2; ++i) {
                int u  = tid + i * 256;           // 512 units of 8 halves
                int r  = u >> 2;
                int c8 = (u & 3) * 8;
                uvec4 v = *reinterpret_cast<const uvec4*>(
                    &A[(size_t)(m0 + r) * K + k0 + c8]);
                *reinterpret_cast<uvec4*>(&As[r * LDA + c8]) = v;
            }
        } else {
            const float* A = (const float*)Aptr;
#pragma unroll
            for (int i = 0; i < 4; ++i) {
                int u  = tid + i * 256;           // 1024 float4 units
                int r  = u >> 3;
                int c4 = (u & 7) * 4;
                float4 v = *reinterpret_cast<const float4*>(
                    &A[(size_t)(m0 + r) * K + k0 + c4]);
                H4 p;
                p.h[0] = (_Float16)v.x; p.h[1] = (_Float16)v.y;
                p.h[2] = (_Float16)v.z; p.h[3] = (_Float16)v.w;
                *reinterpret_cast<uvec2*>(&As[r * LDA + c4]) = p.u;  // 8B store
            }
        }
        // ---- stage W tile, transposed (col-major), packed k-pairs ----
#pragma unroll
        for (int i = 0; i < 2; ++i) {
            int u  = tid + i * 256;               // 512 units: 2 k-rows x 4 cols
            int kr = (u >> 5) * 2;                // 0,2,...,30
            int c4 = (u & 31) * 4;
            float4 v0 = *reinterpret_cast<const float4*>(
                &W[(size_t)(k0 + kr) * N + n0 + c4]);
            float4 v1 = *reinterpret_cast<const float4*>(
                &W[(size_t)(k0 + kr + 1) * N + n0 + c4]);
            H2 p0; p0.h[0] = (_Float16)v0.x; p0.h[1] = (_Float16)v1.x;
            H2 p1; p1.h[0] = (_Float16)v0.y; p1.h[1] = (_Float16)v1.y;
            H2 p2; p2.h[0] = (_Float16)v0.z; p2.h[1] = (_Float16)v1.z;
            H2 p3; p3.h[0] = (_Float16)v0.w; p3.h[1] = (_Float16)v1.w;
            *reinterpret_cast<unsigned*>(&Bs[(c4 + 0) * LDB + kr]) = p0.u;
            *reinterpret_cast<unsigned*>(&Bs[(c4 + 1) * LDB + kr]) = p1.u;
            *reinterpret_cast<unsigned*>(&Bs[(c4 + 2) * LDB + kr]) = p2.u;
            *reinterpret_cast<unsigned*>(&Bs[(c4 + 3) * LDB + kr]) = p3.u;
        }
        __syncthreads();

        // ---- A fragment (16x32 f16 layout, ISA 7.12.2) ----
        Frag16 a;
        {
            int ar = wave * 16 + (lane & 15);
            int kb = (lane < 16) ? 0 : 8;
            a.u[0] = *reinterpret_cast<const uvec4*>(&As[ar * LDA + kb]);
            a.u[1] = *reinterpret_cast<const uvec4*>(&As[ar * LDA + kb + 16]);
        }
        // ---- 8 B fragments + WMMA ----
#pragma unroll
        for (int j = 0; j < 8; ++j) {
            Frag16 bfr;
            int bc  = j * 16 + (lane & 15);
            int kbb = (lane < 16) ? 0 : 8;
            bfr.u[0] = *reinterpret_cast<const uvec4*>(&Bs[bc * LDB + kbb]);
            bfr.u[1] = *reinterpret_cast<const uvec4*>(&Bs[bc * LDB + kbb + 16]);
            acc[j] = wmma_f16(a, bfr, acc[j]);
        }
    }

    // ---- epilogue ----
#pragma unroll
    for (int j = 0; j < 8; ++j) {
        int   col = n0 + j * 16 + (lane & 15);
        float bv  = bias[col];
#pragma unroll
        for (int i = 0; i < 8; ++i) {
            int   row = m0 + wave * 16 + i + ((lane < 16) ? 0 : 8);
            float val = acc[j][i] + bv;
            if (HEAD_OUT) {
                int b = row >> 10, s = row & 1023;   // S = 1024
                int h = col >> 6,  d = col & 63;     // DK = 64
                _Float16* C = (_Float16*)Cptr;
                C[(((size_t)(b * 16 + h)) * 1024 + s) * 64 + d] = (_Float16)val;
            } else {
                float* C = (float*)Cptr;
                C[(size_t)row * N + col] = val;
            }
        }
    }
}

// ---------------------------------------------------------------------------
// Attention core: one block per (b, h, 32-query-row tile).
//   Dynamic LDS (~217 KB of the 320 KB WGP LDS):
//     scores : 32 x 1032 f32  (raw logits, then exp values)
//     pf16   : 32 x 1040 f16  (normalized probs, fragment-ready)
//     qtile  : 32 x 72 f16
//     ktile  : 128 x 72 f16   (reused as V^T 64 x 40 in phase 3)
//   Phase 1: S = Q Kt / 8, mask -> -10000         (WMMA f16)
//   Phase 2: row softmax (wave32 shfl), stream probs f32 to HBM + f16 to LDS
//   Phase 3: ctx = P V                            (WMMA f16)
// ---------------------------------------------------------------------------
__global__ __launch_bounds__(256) void mha_attn_kernel(
    const _Float16* __restrict__ Qh, const _Float16* __restrict__ Kh,
    const _Float16* __restrict__ Vh, const int* __restrict__ mask,
    float* __restrict__ attn_out, _Float16* __restrict__ ctx)
{
    constexpr int SROW = 1032;   // f32 stride (4128B: 16B-aligned, bank-shifted)
    constexpr int PROW = 1040;   // f16 stride (2080B: 16B-aligned, bank-shifted)
    extern __shared__ char smem_raw[];
    float*    scores = (float*)smem_raw;                          // 132096 B
    _Float16* pf16   = (_Float16*)(smem_raw + 132096);            //  66560 B
    _Float16* qtile  = (_Float16*)(smem_raw + 132096 + 66560);    //   4608 B
    _Float16* ktile  = (_Float16*)(smem_raw + 132096 + 66560 + 4608); // 18432 B

    const int tid  = threadIdx.x;
    const int wave = tid >> 5;
    const int lane = tid & 31;

    const int qt = blockIdx.x & 31;        // 32 query tiles of 32 rows
    const int bh = blockIdx.x >> 5;        // 0..127
    const int b  = bh >> 4;
    const size_t headoff = (size_t)bh * 1024 * 64;
    const int q0 = qt * 32;

    // stage Q tile: 32 rows x 64 halves, one 16B unit per thread
    {
        int r = tid >> 3, c8 = (tid & 7) * 8;
        uvec4 v = *reinterpret_cast<const uvec4*>(
            &Qh[headoff + (size_t)(q0 + r) * 64 + c8]);
        *reinterpret_cast<uvec4*>(&qtile[r * 72 + c8]) = v;
    }

    // ---------------- Phase 1: scores = QK^T * 1/8, masked ----------------
    for (int ch = 0; ch < 8; ++ch) {       // 8 chunks of 128 keys
        __syncthreads();
        // stage K chunk: 128 rows x 64 halves (row-major == B operand)
#pragma unroll
        for (int i = 0; i < 4; ++i) {
            int u = tid + i * 256;
            int r = u >> 3, c8 = (u & 7) * 8;
            uvec4 v = *reinterpret_cast<const uvec4*>(
                &Kh[headoff + (size_t)(ch * 128 + r) * 64 + c8]);
            *reinterpret_cast<uvec4*>(&ktile[r * 72 + c8]) = v;
        }
        if (ch < 7)   // prefetch next chunk (global_prefetch_b8)
            __builtin_prefetch(&Kh[headoff + (size_t)((ch + 1) * 128 + (tid >> 3)) * 64], 0, 0);
        __syncthreads();

        v8f acc0 = zero_v8f();   // query rows 0..15
        v8f acc1 = zero_v8f();   // query rows 16..31
#pragma unroll
        for (int ks = 0; ks < 2; ++ks) {   // dk = 0..31, 32..63
            Frag16 bf;
            int bn = wave * 16 + (lane & 15);        // key row within chunk
            int kb = ks * 32 + ((lane < 16) ? 0 : 8);
            bf.u[0] = *reinterpret_cast<const uvec4*>(&ktile[bn * 72 + kb]);
            bf.u[1] = *reinterpret_cast<const uvec4*>(&ktile[bn * 72 + kb + 16]);

            Frag16 a0, a1;
            int ar  = lane & 15;
            int kba = ks * 32 + ((lane < 16) ? 0 : 8);
            a0.u[0] = *reinterpret_cast<const uvec4*>(&qtile[ar * 72 + kba]);
            a0.u[1] = *reinterpret_cast<const uvec4*>(&qtile[ar * 72 + kba + 16]);
            a1.u[0] = *reinterpret_cast<const uvec4*>(&qtile[(16 + ar) * 72 + kba]);
            a1.u[1] = *reinterpret_cast<const uvec4*>(&qtile[(16 + ar) * 72 + kba + 16]);

            acc0 = wmma_f16(a0, bf, acc0);
            acc1 = wmma_f16(a1, bf, acc1);
        }

        int key = ch * 128 + wave * 16 + (lane & 15);
        int mk  = mask[b * 1024 + key];
#pragma unroll
        for (int i = 0; i < 8; ++i) {
            float s0 = acc0[i] * 0.125f;             // 1/sqrt(64)
            float s1 = acc1[i] * 0.125f;
            if (mk == 0) { s0 = -10000.0f; s1 = -10000.0f; }
            int r0 = i + ((lane < 16) ? 0 : 8);
            scores[r0 * SROW + key]        = s0;
            scores[(16 + r0) * SROW + key] = s1;
        }
    }
    __syncthreads();

    // ---------------- Phase 2: softmax; stream f32 probs, stash f16 -------
#pragma unroll
    for (int rr = 0; rr < 4; ++rr) {
        int       r    = wave * 4 + rr;
        float*    row  = &scores[r * SROW];
        _Float16* prow = &pf16[r * PROW];
        float mx = -3.0e38f;
        for (int c = lane; c < 1024; c += 32) mx = fmaxf(mx, row[c]);
#pragma unroll
        for (int off = 16; off; off >>= 1) mx = fmaxf(mx, __shfl_xor(mx, off, 32));
        float sum = 0.0f;
        for (int c = lane; c < 1024; c += 32) {
            float e = __expf(row[c] - mx);
            row[c]  = e;
            sum    += e;
        }
#pragma unroll
        for (int off = 16; off; off >>= 1) sum += __shfl_xor(sum, off, 32);
        float  inv  = 1.0f / sum;   // sum >= 1 always (max subtracted)
        float* gout = &attn_out[((size_t)bh * 1024 + q0 + r) * 1024];
#pragma unroll
        for (int it = 0; it < 16; ++it) {
            int   c  = it * 64 + lane * 2;
            float p0 = row[c] * inv;
            float p1 = row[c + 1] * inv;
            float2 st; st.x = p0; st.y = p1;
            *reinterpret_cast<float2*>(&gout[c]) = st;   // coalesced 8B stores
            H2 pk; pk.h[0] = (_Float16)p0; pk.h[1] = (_Float16)p1;
            *reinterpret_cast<unsigned*>(&prow[c]) = pk.u;
        }
    }

    // ---------------- Phase 3: ctx = P @ V ----------------
    const int mt = wave & 1;        // query-row tile 0/1
    const int ct = wave >> 1;       // dv col tile 0..3
    v8f acc = zero_v8f();

    for (int kt = 0; kt < 32; ++kt) {         // 1024 keys / 32
        __syncthreads();
        // stage V chunk transposed: vtile[d][k_local], stride 40 halves.
        // Thread handles 2 k-rows x 4 d -> packed b32 LDS stores.
        {
            int d4 = (tid & 15) * 4;
            int kr = (tid >> 4) * 2;          // 0,2,...,30
            const _Float16* src0 = &Vh[headoff + (size_t)(kt * 32 + kr) * 64 + d4];
            uvec2 w0 = *reinterpret_cast<const uvec2*>(src0);
            uvec2 w1 = *reinterpret_cast<const uvec2*>(src0 + 64);
            const _Float16* h0 = (const _Float16*)&w0;
            const _Float16* h1 = (const _Float16*)&w1;
#pragma unroll
            for (int j = 0; j < 4; ++j) {
                H2 p; p.h[0] = h0[j]; p.h[1] = h1[j];
                *reinterpret_cast<unsigned*>(&ktile[(d4 + j) * 40 + kr]) = p.u;
            }
        }
        __syncthreads();

        // A fragment straight from fragment-ready f16 probabilities
        Frag16 a;
        {
            int ar = mt * 16 + (lane & 15);
            int kb = kt * 32 + ((lane < 16) ? 0 : 8);
            const _Float16* prow = &pf16[ar * PROW];
            a.u[0] = *reinterpret_cast<const uvec4*>(&prow[kb]);
            a.u[1] = *reinterpret_cast<const uvec4*>(&prow[kb + 16]);
        }
        Frag16 bf;
        {
            int d   = ct * 16 + (lane & 15);
            int kbb = (lane < 16) ? 0 : 8;
            bf.u[0] = *reinterpret_cast<const uvec4*>(&ktile[d * 40 + kbb]);
            bf.u[1] = *reinterpret_cast<const uvec4*>(&ktile[d * 40 + kbb + 16]);
        }
        acc = wmma_f16(a, bf, acc);
    }

    // ctx layout: [B*S, H*DV] row-major, f16
    const int h = bh & 15;
#pragma unroll
    for (int i = 0; i < 8; ++i) {
        int row = mt * 16 + i + ((lane < 16) ? 0 : 8);
        int d   = ct * 16 + (lane & 15);
        ctx[((size_t)(b * 1024) + q0 + row) * 1024 + h * 64 + d] = (_Float16)acc[i];
    }
}

// ---------------------------------------------------------------------------
extern "C" void kernel_launch(void* const* d_in, const int* in_sizes, int n_in,
                              void* d_out, int out_size, void* d_ws, size_t ws_size,
                              hipStream_t stream)
{
    const float* query = (const float*)d_in[0];
    const float* key   = (const float*)d_in[1];
    const float* value = (const float*)d_in[2];
    const int*   mask  = (const int*)d_in[3];
    const float* Wq    = (const float*)d_in[4];
    const float* bq    = (const float*)d_in[5];
    const float* Wk    = (const float*)d_in[6];
    const float* bk    = (const float*)d_in[7];
    const float* Wv    = (const float*)d_in[8];
    const float* bv    = (const float*)d_in[9];
    const float* Wo    = (const float*)d_in[10];
    const float* bo    = (const float*)d_in[11];

    float* out  = (float*)d_out;
    float* attn = out + (size_t)8 * 1024 * 1024;      // [8,16,1024,1024]

    const size_t HEADS_ELEMS = (size_t)8 * 16 * 1024 * 64;   // 8,388,608
    _Float16* Qh  = (_Float16*)d_ws;
    _Float16* Kh  = Qh + HEADS_ELEMS;
    _Float16* Vh  = Kh + HEADS_ELEMS;
    _Float16* ctx = Vh + HEADS_ELEMS;                 // [8192,1024] f16

    const int M = 8192, N = 1024, K = 1024;
    dim3 ggrid(N / 128, M / 128);                     // (8, 64)
    dim3 gblk(256);

    // Q/K/V projections -> f16 head-major
    gemm_wmma_kernel<false, true><<<ggrid, gblk, 0, stream>>>(query, Wq, bq, Qh, M, N, K);
    gemm_wmma_kernel<false, true><<<ggrid, gblk, 0, stream>>>(key,   Wk, bk, Kh, M, N, K);
    gemm_wmma_kernel<false, true><<<ggrid, gblk, 0, stream>>>(value, Wv, bv, Vh, M, N, K);

    // attention: 8*16*32 = 4096 blocks, ~217 KB dynamic LDS each
    const size_t smem = 132096 + 66560 + 4608 + 18432;   // 221,696 B
    mha_attn_kernel<<<dim3(4096), gblk, smem, stream>>>(Qh, Kh, Vh, mask, attn, ctx);

    // output projection -> f32 row-major
    gemm_wmma_kernel<true, false><<<ggrid, gblk, 0, stream>>>(ctx, Wo, bo, out, M, N, K);
}
